// ConeIntersection_48447231099016
// MI455X (gfx1250) — matrix operand order repaired and checked by reading, not compiled, not run on hardware
//
#include <hip/hip_runtime.h>
#include <hip/hip_bf16.h>
#include <math.h>

// ---------------------------------------------------------------------------
// Problem constants (from reference): N=4, B=8192, DIM=1024, HEADS=4, HD=256
// ---------------------------------------------------------------------------
#define N_AX   4
#define BATCH  8192
#define DIMSZ  1024
#define HEADS  4
#define HD     256
#define K1     512          // logits width (2*HD)
#define KH     256          // half of K1; also K of layer 2
#define KS1    16           // K1/32 WMMA k-steps (layer 1)
#define KS2    8            // KH/32 WMMA k-steps (layer 2)
#define TOK    16           // tokens (b,h pairs) per block
#define ROWS   (N_AX * TOK) // 64 rows = 4 WMMA row tiles (one per n)
#define PAD    8            // LDS padding (bf16 elems) to spread banks
#define SLD    (KH + PAD)   // 264-ushort row stride in LDS
#define PI_F   3.14159265358979f

typedef __attribute__((ext_vector_type(16))) __bf16 bf16x16;
typedef __attribute__((ext_vector_type(8)))  float  v8f;

union FragU {
    struct { uint4 lo, hi; } u;
    bf16x16 v;
};

__device__ __forceinline__ unsigned short f2bf(float f) {
    unsigned int x = __float_as_uint(f);
    x += 0x7FFFu + ((x >> 16) & 1u);   // round-to-nearest-even
    return (unsigned short)(x >> 16);
}

__device__ __forceinline__ unsigned int pack2(unsigned short lo, unsigned short hi) {
    return (unsigned int)lo | ((unsigned int)hi << 16);
}

// lo = 8 bf16 at p0, hi = 8 bf16 at p1 (CDNA5 16-bit A/B frag halves)
__device__ __forceinline__ bf16x16 load_frag(const unsigned short* p0,
                                             const unsigned short* p1) {
    FragU f;
    f.u.lo = *(const uint4*)p0;
    f.u.hi = *(const uint4*)p1;
    return f.v;
}

__device__ __forceinline__ v8f wmma_bf16(bf16x16 a, bf16x16 b, v8f c) {
    return __builtin_amdgcn_wmma_f32_16x16x32_bf16(
        /*neg_a=*/false, a, /*neg_b=*/false, b,
        /*c_mod=*/(short)0, c, /*reuse_a=*/false, /*reuse_b=*/false);
}

// ---------------------------------------------------------------------------
// Prep: f32 weights [O=256 x K] -> bf16 in WMMA B-fragment order:
//   out[((ct*KS + ks)*32 + lane)*16 + i] = W[ct*16 + lane%16][ks*32 + 16*(i/8) + 8*(lane/16) + i%8]
// so each B-fragment is one contiguous, wave-coalesced 1 KB block.
// ---------------------------------------------------------------------------
__global__ void cone_prep_wfrag(const float* __restrict__ W,
                                unsigned short* __restrict__ out, int K) {
    const int tid = blockIdx.x * blockDim.x + threadIdx.x;
    if (tid >= HD * K) return;
    const int i    = tid & 15;
    const int lane = (tid >> 4) & 31;
    const int ksct = tid >> 9;          // ct*KS + ks
    const int KS   = K >> 5;
    const int ks   = ksct % KS;
    const int ct   = ksct / KS;
    const int o    = ct * 16 + (lane & 15);
    const int k    = ks * 32 + ((i >> 3) << 4) + ((lane >> 4) << 3) + (i & 7);
    out[tid] = f2bf(W[o * K + k]);
}

// ---------------------------------------------------------------------------
// Fused kernel: logits -> L1 GEMMs -> (softmax / mean) -> L2 GEMMs -> outputs
// ---------------------------------------------------------------------------
__global__ __launch_bounds__(256)
void cone_fused_kernel(const float* __restrict__ axisE, const float* __restrict__ argE,
                       const unsigned short* __restrict__ Wax1, const float* __restrict__ bax1,
                       const unsigned short* __restrict__ War1, const float* __restrict__ bar1,
                       const unsigned short* __restrict__ Wax2, const float* __restrict__ bax2,
                       const unsigned short* __restrict__ War2, const float* __restrict__ bar2,
                       float* __restrict__ axis_out, float* __restrict__ arg_out)
{
    // Single-pass logits: both halves resident (probe of >64KB LDS per WG).
    __shared__ unsigned short s_al[ROWS * SLD];   // 33,792 B; later reused as h1_axis
    __shared__ unsigned short s_ah[ROWS * SLD];   // 33,792 B
    __shared__ unsigned short s_mean[TOK * SLD];  //  8,448 B

    const int tid  = threadIdx.x;
    const int wave = tid >> 5;
    const int lane = tid & 31;
    const int c0   = wave * 2;               // first of this wave's 2 column tiles
    const int tok0 = blockIdx.x * TOK;       // token = b*HEADS + h; elem = tok*HD + k
    const size_t NB = (size_t)BATCH * DIMSZ; // per-n stride in elements

    const v8f vzero = {0.f, 0.f, 0.f, 0.f, 0.f, 0.f, 0.f, 0.f};

    // ---------------- Phase 0: build al/ah (bf16) in one pass ----------------
    for (int idx = tid; idx < ROWS * (HD / 4); idx += 256) {
        const int r  = idx >> 6;              // 64 float4 per row
        const int k4 = (idx & 63) << 2;
        const int n  = r >> 4, t = r & 15;
        const size_t g = (size_t)n * NB + (size_t)(tok0 + t) * HD + k4;
        const float4 a4 = *(const float4*)(axisE + g);
        const float4 g4 = *(const float4*)(argE + g);
        unsigned int* dl = (unsigned int*)&s_al[r * SLD + k4];
        unsigned int* dh = (unsigned int*)&s_ah[r * SLD + k4];
        dl[0] = pack2(f2bf(fmaf(-0.5f, g4.x, a4.x)), f2bf(fmaf(-0.5f, g4.y, a4.y)));
        dl[1] = pack2(f2bf(fmaf(-0.5f, g4.z, a4.z)), f2bf(fmaf(-0.5f, g4.w, a4.w)));
        dh[0] = pack2(f2bf(fmaf( 0.5f, g4.x, a4.x)), f2bf(fmaf( 0.5f, g4.y, a4.y)));
        dh[1] = pack2(f2bf(fmaf( 0.5f, g4.z, a4.z)), f2bf(fmaf( 0.5f, g4.w, a4.w)));
    }
    __syncthreads();

    // ------------- Layer 1: D = relu([al|ah] x W1^T + b1), 2 sets -------------
    v8f accx[N_AX][2], accr[N_AX][2];
#pragma unroll
    for (int n = 0; n < N_AX; ++n)
#pragma unroll
        for (int c = 0; c < 2; ++c) { accx[n][c] = vzero; accr[n][c] = vzero; }

    for (int ks = 0; ks < KS1; ++ks) {
        const unsigned short* sbuf = (ks < KS2) ? s_al : s_ah;
        const int k0 = (ks & (KS2 - 1)) * 32;
        const int kA = k0 + ((lane >> 4) << 3);
        bf16x16 aF[N_AX];
#pragma unroll
        for (int n = 0; n < N_AX; ++n) {
            const unsigned short* rp = sbuf + (n * 16 + (lane & 15)) * SLD;
            aF[n] = load_frag(rp + kA, rp + kA + 16);
        }
#pragma unroll
        for (int c = 0; c < 2; ++c) {
            const size_t fo = (((size_t)(c0 + c) * KS1 + ks) * 32 + lane) * 16;
            const bf16x16 bx = load_frag(Wax1 + fo, Wax1 + fo + 8);
            const bf16x16 br = load_frag(War1 + fo, War1 + fo + 8);
#pragma unroll
            for (int n = 0; n < N_AX; ++n) {
                accx[n][c] = wmma_bf16(aF[n], bx, accx[n][c]);
                accr[n][c] = wmma_bf16(aF[n], br, accr[n][c]);
            }
        }
    }
    __syncthreads();   // all waves done reading s_al/s_ah

    // bias + relu; h1_axis -> s_al (reuse); mean_n relu(arg_h1) -> s_mean
#pragma unroll
    for (int c = 0; c < 2; ++c) {
        const int col = (c0 + c) * 16 + (lane & 15);
        const float bA = bax1[col];
        const float bR = bar1[col];
        float msum[8];
#pragma unroll
        for (int j = 0; j < 8; ++j) msum[j] = 0.0f;
#pragma unroll
        for (int n = 0; n < N_AX; ++n)
#pragma unroll
            for (int j = 0; j < 8; ++j) {
                const int t = j + ((lane >> 4) << 3);     // C layout: M = j (+8 hi lanes)
                float va = fmaxf(accx[n][c][j] + bA, 0.0f);
                float vr = fmaxf(accr[n][c][j] + bR, 0.0f);
                s_al[(n * 16 + t) * SLD + col] = f2bf(va);
                msum[j] += vr;
            }
#pragma unroll
        for (int j = 0; j < 8; ++j) {
            const int t = j + ((lane >> 4) << 3);
            s_mean[t * SLD + col] = f2bf(msum[j] * 0.25f);
        }
    }
    __syncthreads();

    // ---------------- Layer 2: axis logits + arg gate GEMMs ----------------
    v8f acc2[N_AX][2], accg[2];
#pragma unroll
    for (int n = 0; n < N_AX; ++n)
#pragma unroll
        for (int c = 0; c < 2; ++c) acc2[n][c] = vzero;
    accg[0] = vzero; accg[1] = vzero;

    for (int ks = 0; ks < KS2; ++ks) {
        const int k0 = ks * 32;
        const int kA = k0 + ((lane >> 4) << 3);
        bf16x16 aF[N_AX];
#pragma unroll
        for (int n = 0; n < N_AX; ++n) {
            const unsigned short* rp = s_al + (n * 16 + (lane & 15)) * SLD;
            aF[n] = load_frag(rp + kA, rp + kA + 16);
        }
        const unsigned short* mp = s_mean + (lane & 15) * SLD;
        const bf16x16 aM = load_frag(mp + kA, mp + kA + 16);
#pragma unroll
        for (int c = 0; c < 2; ++c) {
            const size_t fo = (((size_t)(c0 + c) * KS2 + ks) * 32 + lane) * 16;
            const bf16x16 b2 = load_frag(Wax2 + fo, Wax2 + fo + 8);
            const bf16x16 bg = load_frag(War2 + fo, War2 + fo + 8);
#pragma unroll
            for (int n = 0; n < N_AX; ++n)
                acc2[n][c] = wmma_bf16(aF[n], b2, acc2[n][c]);
            accg[c] = wmma_bf16(aM, bg, accg[c]);
        }
    }

    // ------ epilogue: softmax_n, circular mean, atan fixup; gate*min_n ------
#pragma unroll
    for (int c = 0; c < 2; ++c) {
        const int col = (c0 + c) * 16 + (lane & 15);
        const float b2 = bax2[col];
        const float bg = bar2[col];
#pragma unroll
        for (int j = 0; j < 8; ++j) {
            const int t = j + ((lane >> 4) << 3);
            const size_t gpos = (size_t)(tok0 + t) * HD + col;   // == b*DIM + h*HD + col

            // softmax over n (4 values live in 4 accumulator tiles)
            float l0 = acc2[0][c][j] + b2, l1 = acc2[1][c][j] + b2;
            float l2 = acc2[2][c][j] + b2, l3 = acc2[3][c][j] + b2;
            const float m = fmaxf(fmaxf(l0, l1), fmaxf(l2, l3));
            const float e0 = __expf(l0 - m), e1 = __expf(l1 - m);
            const float e2 = __expf(l2 - m), e3 = __expf(l3 - m);
            const float inv = 1.0f / (e0 + e1 + e2 + e3);

            const float a0 = axisE[gpos],          a1 = axisE[gpos + NB];
            const float a2 = axisE[gpos + 2 * NB], a3 = axisE[gpos + 3 * NB];
            float x = (e0 * __cosf(a0) + e1 * __cosf(a1) +
                       e2 * __cosf(a2) + e3 * __cosf(a3)) * inv;
            const float y = (e0 * __sinf(a0) + e1 * __sinf(a1) +
                             e2 * __sinf(a2) + e3 * __sinf(a3)) * inv;
            if (fabsf(x) < 0.001f) x = 0.001f;
            float axv = atanf(y / x);
            if (x < 0.0f) axv += (y < 0.0f) ? -PI_F : PI_F;
            axis_out[gpos] = axv;

            // arg branch: sigmoid gate * min_n(arg)
            float gv = accg[c][j] + bg;
            gv = 1.0f / (1.0f + __expf(-gv));
            const float r0 = argE[gpos],          r1 = argE[gpos + NB];
            const float r2 = argE[gpos + 2 * NB], r3 = argE[gpos + 3 * NB];
            arg_out[gpos] = fminf(fminf(r0, r1), fminf(r2, r3)) * gv;
        }
    }
}

// ---------------------------------------------------------------------------
// Host entry
// ---------------------------------------------------------------------------
extern "C" void kernel_launch(void* const* d_in, const int* in_sizes, int n_in,
                              void* d_out, int out_size, void* d_ws, size_t ws_size,
                              hipStream_t stream) {
    (void)in_sizes; (void)n_in; (void)out_size; (void)ws_size;
    const float* axisE = (const float*)d_in[0];
    const float* argE  = (const float*)d_in[1];
    const float* Wax1f = (const float*)d_in[2];
    const float* bax1  = (const float*)d_in[3];
    const float* War1f = (const float*)d_in[4];
    const float* bar1  = (const float*)d_in[5];
    const float* Wax2f = (const float*)d_in[6];
    const float* bax2  = (const float*)d_in[7];
    const float* War2f = (const float*)d_in[8];
    const float* bar2  = (const float*)d_in[9];

    // bf16 fragment-ordered weight staging in workspace (786 KB)
    unsigned short* wax1 = (unsigned short*)d_ws;
    unsigned short* war1 = wax1 + HD * K1;     // 131072 elems each
    unsigned short* wax2 = war1 + HD * K1;
    unsigned short* war2 = wax2 + HD * KH;     //  65536 elems each

    cone_prep_wfrag<<<(HD * K1 + 255) / 256, 256, 0, stream>>>(Wax1f, wax1, K1);
    cone_prep_wfrag<<<(HD * K1 + 255) / 256, 256, 0, stream>>>(War1f, war1, K1);
    cone_prep_wfrag<<<(HD * KH + 255) / 256, 256, 0, stream>>>(Wax2f, wax2, KH);
    cone_prep_wfrag<<<(HD * KH + 255) / 256, 256, 0, stream>>>(War2f, war2, KH);

    float* axis_out = (float*)d_out;
    float* arg_out  = axis_out + (size_t)BATCH * DIMSZ;

    const int nblocks = (BATCH * HEADS) / TOK;   // 2048
    cone_fused_kernel<<<nblocks, 256, 0, stream>>>(
        axisE, argE, wax1, bax1, war1, bar1, wax2, bax2, war2, bar2,
        axis_out, arg_out);
}
